// SpanPrunerMangoes_3444563771561
// MI455X (gfx1250) — matrix operand order repaired and checked by reading, not compiled, not run on hardware
//
#include <hip/hip_runtime.h>
#include <hip/hip_bf16.h>
#include <math.h>

// ---------------- problem constants ----------------
#define NUM_TOKENS 2048
#define HIDDEN 768
#define MAX_SPAN_WIDTH 20
#define NUM_CAND (NUM_TOKENS * MAX_SPAN_WIDTH)   // 40960
#define FEAT 20
#define UNARY 1024
#define SPAN_DIM (3 * HIDDEN + FEAT)             // 2324
#define NTOP 512
#define K_PAD 2336                                // 73*32
#define KC 292                                    // K_PAD/8 (8-elem bf16 chunks)
#define K32 73
#define LDSTRIDE 33                               // padded A-LDS stride (uint4 units)
#define SORT_N 65536
#define SORT_TILE 4096

typedef __bf16 bf16_t;
typedef bf16_t v16bf __attribute__((ext_vector_type(16)));
typedef float  v8f   __attribute__((ext_vector_type(8)));

union Frag { uint4 q[2]; v16bf v; };

__device__ __forceinline__ unsigned bf16bits(float f) {
    unsigned u = __float_as_uint(f);
    return ((u + 0x7fffu + ((u >> 16) & 1u)) >> 16) & 0xffffu;
}

// ---------------- tok_att = hidden @ w_attn + b_attn ----------------
__global__ __launch_bounds__(256)
void tok_att_kernel(const float* __restrict__ hidden, const float* __restrict__ w_attn,
                    const float* __restrict__ b_attn, float* __restrict__ tok_att) {
    int wave = threadIdx.x >> 5, lane = threadIdx.x & 31;
    int t = blockIdx.x * 8 + wave;
    const float* row = hidden + (size_t)t * HIDDEN;
    float s = 0.f;
    for (int i = lane; i < HIDDEN; i += 32) s += row[i] * w_attn[i];
    s += __shfl_xor(s, 1);  s += __shfl_xor(s, 2);
    s += __shfl_xor(s, 4);  s += __shfl_xor(s, 8);
    s += __shfl_xor(s, 16);
    if (lane == 0) tok_att[t] = s + b_attn[0];
}

// ---------------- width prior scores (20 values, fold bp2) ----------------
__global__ __launch_bounds__(256)
void prior_kernel(const float* __restrict__ ewp, const float* __restrict__ Wp1,
                  const float* __restrict__ bp1, const float* __restrict__ Wp2,
                  const float* __restrict__ bp2, float* __restrict__ prior) {
    __shared__ float acc[MAX_SPAN_WIDTH];
    if (threadIdx.x < MAX_SPAN_WIDTH) acc[threadIdx.x] = 0.f;
    __syncthreads();
    for (int idx = threadIdx.x; idx < MAX_SPAN_WIDTH * UNARY; idx += 256) {
        int w = idx >> 10, n = idx & (UNARY - 1);
        float v = bp1[n];
        for (int f = 0; f < FEAT; ++f) v += ewp[w * FEAT + f] * Wp1[f * UNARY + n];
        v = v > 0.f ? v : 0.f;
        atomicAdd(&acc[w], v * Wp2[n]);
    }
    __syncthreads();
    if (threadIdx.x < MAX_SPAN_WIDTH) prior[threadIdx.x] = acc[threadIdx.x] + bp2[0];
}

// ---------------- W1 fp32 -> bf16 chunked [kc][n] (16B per chunk) ----------------
__global__ __launch_bounds__(256)
void w1_convert_kernel(const float* __restrict__ W1, uint4* __restrict__ W1p) {
    int idx = blockIdx.x * 256 + threadIdx.x;
    if (idx >= KC * UNARY) return;
    int kc = idx >> 10, n = idx & (UNARY - 1);
    unsigned o[4];
    for (int j = 0; j < 4; ++j) {
        int k0 = kc * 8 + 2 * j;
        float f0 = (k0     < SPAN_DIM) ? W1[(size_t)k0 * UNARY + n]       : 0.f;
        float f1 = (k0 + 1 < SPAN_DIM) ? W1[(size_t)(k0 + 1) * UNARY + n] : 0.f;
        o[j] = (bf16bits(f1) << 16) | bf16bits(f0);
    }
    W1p[idx] = make_uint4(o[0], o[1], o[2], o[3]);
}

// ---------------- span_emb build + masked softmax attention ----------------
__global__ __launch_bounds__(256)
void span_emb_kernel(const float* __restrict__ hidden, const int* __restrict__ cstarts,
                     const int* __restrict__ cends, const float* __restrict__ emb_width,
                     const float* __restrict__ tok_att, float* __restrict__ out) {
    int wave = threadIdx.x >> 5, lane = threadIdx.x & 31;
    int c = blockIdx.x * 8 + wave;
    int s = cstarts[c], e = cends[c], w = e - s;
    float a = (lane <= w) ? tok_att[s + lane] : -INFINITY;
    float m = a;
    m = fmaxf(m, __shfl_xor(m, 1));  m = fmaxf(m, __shfl_xor(m, 2));
    m = fmaxf(m, __shfl_xor(m, 4));  m = fmaxf(m, __shfl_xor(m, 8));
    m = fmaxf(m, __shfl_xor(m, 16));
    float ex = (lane <= w) ? __expf(a - m) : 0.f;
    float sum = ex;
    sum += __shfl_xor(sum, 1);  sum += __shfl_xor(sum, 2);
    sum += __shfl_xor(sum, 4);  sum += __shfl_xor(sum, 8);
    sum += __shfl_xor(sum, 16);
    float wgt = ex / sum;

    float* dst = out + (size_t)c * SPAN_DIM;
    const float4* hs = (const float4*)(hidden + (size_t)s * HIDDEN);
    const float4* he = (const float4*)(hidden + (size_t)e * HIDDEN);
    float4* d0 = (float4*)dst;
    float4* d1 = (float4*)(dst + HIDDEN);
    for (int i = lane; i < HIDDEN / 4; i += 32) { d0[i] = hs[i]; d1[i] = he[i]; }
    if (lane < FEAT) dst[2 * HIDDEN + lane] = emb_width[w * FEAT + lane];

    float4 acc[6];
    for (int ii = 0; ii < 6; ++ii) acc[ii] = make_float4(0.f, 0.f, 0.f, 0.f);
    for (int t = 0; t <= w; ++t) {
        float wt = __shfl(wgt, t);
        const float4* hr = (const float4*)(hidden + (size_t)(s + t) * HIDDEN);
        for (int ii = 0; ii < 6; ++ii) {
            float4 h4 = hr[lane + 32 * ii];
            acc[ii].x += wt * h4.x; acc[ii].y += wt * h4.y;
            acc[ii].z += wt * h4.z; acc[ii].w += wt * h4.w;
        }
    }
    float4* d2 = (float4*)(dst + 2 * HIDDEN + FEAT);
    for (int ii = 0; ii < 6; ++ii) d2[lane + 32 * ii] = acc[ii];
}

// ---------------- fused WMMA GEMM + relu + W2 dot -> span scores ----------------
// 2M x 2N register tile per wave job: 4 WMMA per (4 global + 4 LDS) b128 loads.
__global__ __launch_bounds__(256)
void gemm_score_kernel(const float* __restrict__ spanEmb, const uint4* __restrict__ W1p,
                       const float* __restrict__ b1, const float* __restrict__ W2,
                       const float* __restrict__ b2, const float* __restrict__ prior,
                       const int* __restrict__ cstarts, const int* __restrict__ cends,
                       float* __restrict__ scores) {
    __shared__ uint4 Alds[KC * LDSTRIDE];   // [kc][m] bf16 8-chunks, 32 rows, padded stride
    __shared__ float lds_score[32];
    int tid = threadIdx.x;
    int m0 = blockIdx.x * 32;

    // Stage A block (32 rows) fp32 -> bf16 chunks in LDS. 8 threads per row.
    {
        int m = tid >> 3, kcb = tid & 7;
        const float* row = spanEmb + (size_t)(m0 + m) * SPAN_DIM;
        for (int kc = kcb; kc < KC; kc += 8) {
            int k = kc * 8;
            float f[8];
            if (k + 8 <= SPAN_DIM) {
                const float4* p = (const float4*)(row + k);
                float4 x = p[0], y = p[1];
                f[0] = x.x; f[1] = x.y; f[2] = x.z; f[3] = x.w;
                f[4] = y.x; f[5] = y.y; f[6] = y.z; f[7] = y.w;
            } else {
                for (int j = 0; j < 8; ++j) f[j] = (k + j < SPAN_DIM) ? row[k + j] : 0.f;
            }
            uint4 o;
            unsigned* op = (unsigned*)&o;
            for (int j = 0; j < 4; ++j)
                op[j] = (bf16bits(f[2 * j + 1]) << 16) | bf16bits(f[2 * j]);
            Alds[kc * LDSTRIDE + m] = o;
        }
    }
    if (tid < 32) lds_score[tid] = 0.f;
    __syncthreads();

    int wave = tid >> 5, lane = tid & 31;
    int lhalf = (lane < 16) ? 0 : 1;
    int lm = lane & 15;
    float scoreAcc[16];
    for (int i = 0; i < 16; ++i) scoreAcc[i] = 0.f;

    // 32 N-tile-pairs over UNARY=1024; each wave takes 4 pairs.
    for (int ntp = wave; ntp < UNARY / 32; ntp += 8) {
        int col0 = ntp * 32 + lm;
        int col1 = col0 + 16;
        const uint4* bp0 = W1p + col0;
        const uint4* bp1v = W1p + col1;
        v8f a00 = {}, a01 = {}, a10 = {}, a11 = {};
        for (int k32 = 0; k32 < K32; ++k32) {
            int kc0 = k32 * 4;
            // B: lane<16 holds K 0..15 (chunks kc0,kc0+1); lane>=16 K 16..31
            int bkc = kc0 + lhalf * 2;
            Frag bF0, bF1;
            bF0.q[0] = bp0[bkc * UNARY];
            bF0.q[1] = bp0[(bkc + 1) * UNARY];
            bF1.q[0] = bp1v[bkc * UNARY];
            bF1.q[1] = bp1v[(bkc + 1) * UNARY];
            // A: lane<16 holds K 0..15 (chunks kc0, kc0+1); lane>=16 K 16..31
            int akc = kc0 + lhalf * 2;
            Frag aF0, aF1;
            aF0.q[0] = Alds[akc * LDSTRIDE + lm];
            aF0.q[1] = Alds[(akc + 1) * LDSTRIDE + lm];
            aF1.q[0] = Alds[akc * LDSTRIDE + 16 + lm];
            aF1.q[1] = Alds[(akc + 1) * LDSTRIDE + 16 + lm];
            a00 = __builtin_amdgcn_wmma_f32_16x16x32_bf16(false, aF0.v, false, bF0.v,
                                                          (short)0, a00, false, false);
            a10 = __builtin_amdgcn_wmma_f32_16x16x32_bf16(false, aF1.v, false, bF0.v,
                                                          (short)0, a10, false, false);
            a01 = __builtin_amdgcn_wmma_f32_16x16x32_bf16(false, aF0.v, false, bF1.v,
                                                          (short)0, a01, false, false);
            a11 = __builtin_amdgcn_wmma_f32_16x16x32_bf16(false, aF1.v, false, bF1.v,
                                                          (short)0, a11, false, false);
        }
        float b1c0 = b1[col0], w2c0 = W2[col0];
        float b1c1 = b1[col1], w2c1 = W2[col1];
        for (int v = 0; v < 8; ++v) {
            float h00 = a00[v] + b1c0; h00 = h00 > 0.f ? h00 : 0.f;
            float h01 = a01[v] + b1c1; h01 = h01 > 0.f ? h01 : 0.f;
            float h10 = a10[v] + b1c0; h10 = h10 > 0.f ? h10 : 0.f;
            float h11 = a11[v] + b1c1; h11 = h11 > 0.f ? h11 : 0.f;
            scoreAcc[v]     += h00 * w2c0 + h01 * w2c1;
            scoreAcc[8 + v] += h10 * w2c0 + h11 * w2c1;
        }
    }
    // lanes 0..15 share rows (v + 0/8 per half); reduce each half, then LDS atomic
    for (int i = 0; i < 16; ++i) {
        float s = scoreAcc[i];
        s += __shfl_xor(s, 1); s += __shfl_xor(s, 2);
        s += __shfl_xor(s, 4); s += __shfl_xor(s, 8);
        if (lm == 0) {
            int row = (i >> 3) * 16 + (i & 7) + lhalf * 8;
            atomicAdd(&lds_score[row], s);
        }
    }
    __syncthreads();
    if (tid < 32) {
        int c = m0 + tid;
        int w = cends[c] - cstarts[c];
        scores[c] = lds_score[tid] + b2[0] + prior[w];
    }
}

// ---------------- sort keys: descending score, ties -> smaller idx first ----------------
__global__ __launch_bounds__(256)
void key_init_kernel(const float* __restrict__ scores, unsigned long long* __restrict__ keys) {
    int i = blockIdx.x * 256 + threadIdx.x;
    if (i >= SORT_N) return;
    if (i < NUM_CAND) {
        unsigned u = __float_as_uint(scores[i]);
        u = (u & 0x80000000u) ? ~u : (u | 0x80000000u);
        keys[i] = ((unsigned long long)u << 32) | (unsigned)(~(unsigned)i);
    } else {
        keys[i] = 0ull;   // smaller than any real key -> ends up last (descending)
    }
}

__global__ __launch_bounds__(1024)
void bitonic_local_sort(unsigned long long* __restrict__ keys) {
    __shared__ unsigned long long sk[SORT_TILE];
    int base = blockIdx.x * SORT_TILE;
    for (int t = threadIdx.x; t < SORT_TILE; t += 1024) sk[t] = keys[base + t];
    __syncthreads();
    for (int k = 2; k <= SORT_TILE; k <<= 1) {
        for (int j = k >> 1; j > 0; j >>= 1) {
            for (int p = threadIdx.x; p < SORT_TILE / 2; p += 1024) {
                int ii = (p / j) * (j << 1) + (p % j);
                int pp = ii + j;
                bool desc = (((base + ii) & k) == 0);   // overall descending sort
                unsigned long long a = sk[ii], b = sk[pp];
                if ((a < b) == desc) { sk[ii] = b; sk[pp] = a; }
            }
            __syncthreads();
        }
    }
    for (int t = threadIdx.x; t < SORT_TILE; t += 1024) keys[base + t] = sk[t];
}

__global__ __launch_bounds__(1024)
void bitonic_global_step(unsigned long long* __restrict__ keys, int j, int k) {
    int p = blockIdx.x * 1024 + threadIdx.x;      // SORT_N/2 pairs
    int ii = (p / j) * (j << 1) + (p % j);
    int pp = ii + j;
    bool desc = ((ii & k) == 0);
    unsigned long long a = keys[ii], b = keys[pp];
    if ((a < b) == desc) { keys[ii] = b; keys[pp] = a; }
}

__global__ __launch_bounds__(1024)
void bitonic_local_merge(unsigned long long* __restrict__ keys, int k) {
    __shared__ unsigned long long sk[SORT_TILE];
    int base = blockIdx.x * SORT_TILE;
    for (int t = threadIdx.x; t < SORT_TILE; t += 1024) sk[t] = keys[base + t];
    __syncthreads();
    for (int j = SORT_TILE >> 1; j > 0; j >>= 1) {
        for (int p = threadIdx.x; p < SORT_TILE / 2; p += 1024) {
            int ii = (p / j) * (j << 1) + (p % j);
            int pp = ii + j;
            bool desc = (((base + ii) & k) == 0);
            unsigned long long a = sk[ii], b = sk[pp];
            if ((a < b) == desc) { sk[ii] = b; sk[pp] = a; }
        }
        __syncthreads();
    }
    for (int t = threadIdx.x; t < SORT_TILE; t += 1024) keys[base + t] = sk[t];
}

// ---------------- sequential greedy NMS scan + final (start,end) sort ----------------
__global__ __launch_bounds__(32)
void scan_kernel(const unsigned long long* __restrict__ keys,
                 const int* __restrict__ cstarts, const int* __restrict__ cends,
                 const int* __restrict__ speaker, const float* __restrict__ scores,
                 float* __restrict__ outIdx, float* __restrict__ outStarts,
                 float* __restrict__ outEnds, float* __restrict__ outScores,
                 float* __restrict__ outSpk, int* __restrict__ acc_idx_ws) {
    __shared__ int LE[NUM_TOKENS];
    __shared__ int ES[NUM_TOKENS];
    __shared__ int accIdx[NTOP];
    __shared__ unsigned long long skey[NTOP];
    int lane = threadIdx.x;
    for (int t = lane; t < NUM_TOKENS; t += 32) { LE[t] = -1; ES[t] = NUM_TOKENS; }
    __syncthreads();
    int count = 0;
    for (int r = 0; r < NUM_CAND && count < NTOP; ++r) {
        unsigned long long kk = keys[r];
        int i = (int)(~(unsigned)(kk & 0xffffffffull));
        int s = cstarts[i], e = cends[i];
        int wspan = e - s;
        bool c1 = (lane < wspan) && (LE[s + 1 + lane] > e);
        bool c2 = (lane < wspan) && (ES[s + lane] < s);
        bool cross = (__ballot(c1 || c2) != 0ull);
        if (!cross) {
            if (lane == 0) {
                LE[s] = max(LE[s], e);
                ES[e] = min(ES[e], s);
                accIdx[count] = i;
                skey[count] = ((unsigned long long)(s * (NUM_TOKENS + 1) + e) << 16)
                              | (unsigned)count;
            }
            ++count;
        }
        __syncthreads();
    }
    for (int t = lane + count; t < NTOP; t += 32) skey[t] = ~0ull;
    __syncthreads();
    // ascending bitonic on (start,end,seq) keys
    for (int k = 2; k <= NTOP; k <<= 1) {
        for (int j = k >> 1; j > 0; j >>= 1) {
            for (int p = lane; p < NTOP / 2; p += 32) {
                int ii = (p / j) * (j << 1) + (p % j);
                int pp = ii + j;
                bool asc = ((ii & k) == 0);
                unsigned long long a = skey[ii], b = skey[pp];
                if ((a > b) == asc) { skey[ii] = b; skey[pp] = a; }
            }
            __syncthreads();
        }
    }
    for (int r = lane; r < NTOP; r += 32) {
        int seq = (r < count) ? (int)(skey[r] & 0xffffull) : (int)(skey[0] & 0xffffull);
        int ci = accIdx[seq];
        int s = cstarts[ci], e = cends[ci];
        outIdx[r]    = (float)ci;
        outStarts[r] = (float)s;
        outEnds[r]   = (float)e;
        outScores[r] = scores[ci];
        outSpk[r]    = (float)speaker[s];
        acc_idx_ws[r] = ci;
    }
}

// ---------------- gather pruned span embeddings ----------------
__global__ __launch_bounds__(256)
void gather_emb_kernel(const float* __restrict__ spanEmb, const int* __restrict__ acc_idx,
                       float* __restrict__ outEmb) {
    int r = blockIdx.x;
    int ci = acc_idx[r];
    const float4* src = (const float4*)(spanEmb + (size_t)ci * SPAN_DIM);
    float4* dst = (float4*)(outEmb + (size_t)r * SPAN_DIM);
    for (int i = threadIdx.x; i < SPAN_DIM / 4; i += 256) dst[i] = src[i];
}

// ---------------- launcher ----------------
extern "C" void kernel_launch(void* const* d_in, const int* in_sizes, int n_in,
                              void* d_out, int out_size, void* d_ws, size_t ws_size,
                              hipStream_t stream) {
    (void)in_sizes; (void)n_in; (void)out_size; (void)ws_size;
    const float* hidden   = (const float*)d_in[0];
    const int*   cstarts  = (const int*)d_in[1];
    const int*   cends    = (const int*)d_in[2];
    const int*   speaker  = (const int*)d_in[3];
    const float* w_attn   = (const float*)d_in[4];
    const float* b_attn   = (const float*)d_in[5];
    const float* W1       = (const float*)d_in[6];
    const float* b1       = (const float*)d_in[7];
    const float* W2       = (const float*)d_in[8];
    const float* b2       = (const float*)d_in[9];
    const float* embw     = (const float*)d_in[10];
    const float* embwp    = (const float*)d_in[11];
    const float* Wp1      = (const float*)d_in[12];
    const float* bp1      = (const float*)d_in[13];
    const float* Wp2      = (const float*)d_in[14];
    const float* bp2      = (const float*)d_in[15];

    char* ws = (char*)d_ws;
    uint4* W1p                = (uint4*)ws;                                   // 4,784,128 B
    float* tok_att            = (float*)(ws + 4784128);                       // 8 KB
    float* prior              = (float*)(ws + 4792320);                       // 128 B
    float* scores             = (float*)(ws + 4792448);                       // 160 KB
    unsigned long long* keys  = (unsigned long long*)(ws + 4956288);          // 512 KB
    int*   acc_idx            = (int*)(ws + 5480576);                         // 2 KB

    float* out       = (float*)d_out;
    float* outSpan   = out;                                  // 40960*2324
    float* outIdx    = out + (size_t)NUM_CAND * SPAN_DIM;    // 512
    float* outStarts = outIdx + NTOP;
    float* outEnds   = outStarts + NTOP;
    float* outEmb    = outEnds + NTOP;                       // 512*2324
    float* outScores = outEmb + (size_t)NTOP * SPAN_DIM;
    float* outSpk    = outScores + NTOP;

    tok_att_kernel<<<NUM_TOKENS / 8, 256, 0, stream>>>(hidden, w_attn, b_attn, tok_att);
    prior_kernel<<<1, 256, 0, stream>>>(embwp, Wp1, bp1, Wp2, bp2, prior);
    w1_convert_kernel<<<(KC * UNARY) / 256, 256, 0, stream>>>(W1, W1p);
    span_emb_kernel<<<NUM_CAND / 8, 256, 0, stream>>>(hidden, cstarts, cends, embw,
                                                      tok_att, outSpan);
    gemm_score_kernel<<<NUM_CAND / 32, 256, 0, stream>>>(outSpan, W1p, b1, W2, b2, prior,
                                                         cstarts, cends, scores);
    key_init_kernel<<<SORT_N / 256, 256, 0, stream>>>(scores, keys);
    bitonic_local_sort<<<SORT_N / SORT_TILE, 1024, 0, stream>>>(keys);
    for (int k = SORT_TILE * 2; k <= SORT_N; k <<= 1) {
        for (int j = k >> 1; j > SORT_TILE / 2; j >>= 1)
            bitonic_global_step<<<SORT_N / 2048, 1024, 0, stream>>>(keys, j, k);
        bitonic_local_merge<<<SORT_N / SORT_TILE, 1024, 0, stream>>>(keys, k);
    }
    scan_kernel<<<1, 32, 0, stream>>>(keys, cstarts, cends, speaker, scores,
                                      outIdx, outStarts, outEnds, outScores, outSpk, acc_idx);
    gather_emb_kernel<<<NTOP, 256, 0, stream>>>(outSpan, acc_idx, outEmb);
}
